// GATNE_I_54863912239176
// MI455X (gfx1250) — matrix operand order repaired and checked by reading, not compiled, not run on hardware
//
#include <hip/hip_runtime.h>
#include <hip/hip_bf16.h>

// ---------------------------------------------------------------------------
// GATNE-I fused embedding kernel for MI455X (gfx1250, wave32, WMMA).
//
// Roofline: dominant cost is streaming nbr_features (1.07 GB) once ->
// ~46 us @ 23.3 TB/s. Mean-over-samples is hoisted before the edge GEMM,
// shrinking matrix FLOPs 10x to ~8.6 GFLOP, executed as bf16
// v_wmma_f32_16x16x32_bf16 with fp32 accumulation (~us-scale).
// One block = 16 nodes (WMMA M=16), fully fused through softmax-attention
// and final L2 normalization.
// ---------------------------------------------------------------------------

typedef __attribute__((ext_vector_type(16))) __bf16 v16bf;
typedef __attribute__((ext_vector_type(8)))  __bf16 v8bf;
typedef __attribute__((ext_vector_type(8)))  float  v8f;

#define FDIM   256   // feature dim (K of big GEMMs)
#define TNUM   4     // edge types
#define SNUM   10    // neighbor samples
#define DDIM   64    // edge embedding size
#define ADIM   32    // attention dim
#define EDIM   128   // output embedding size
#define NPB    16    // nodes per block (== WMMA M)
#define APAD   264   // padded LDS row (256 + 8) -> conflict-free ds_load_b128

__device__ __forceinline__ unsigned short bfbits(float x) {
  unsigned u = __builtin_bit_cast(unsigned, x);
  u += 0x7FFFu + ((u >> 16) & 1u);           // round-to-nearest-even
  return (unsigned short)(u >> 16);
}
__device__ __forceinline__ unsigned pk2(float x, float y) {
  return (unsigned)bfbits(x) | ((unsigned)bfbits(y) << 16);
}

// ---------------------------------------------------------------------------
// Weight pre-pack: lay weights out in the exact WMMA B-fragment order so the
// main kernel's B load is one contiguous 32B global load per lane (L2-hot).
// B order chosen here (k contiguous, as in the A load): lane L holds N = L%16, K = kb + (L/16)*16 + e.
// ---------------------------------------------------------------------------
__global__ void pack_edge_kernel(const float* __restrict__ W,  // [4][256][64]
                                 unsigned short* __restrict__ out) {
  int p    = blockIdx.x * blockDim.x + threadIdx.x;  // 65536 total
  int e    = p & 15;
  int lane = (p >> 4) & 31;
  int frag = p >> 9;                                  // 0..127
  int ks = frag & 7, dt = (frag >> 3) & 3, t = frag >> 5;
  int K  = ks * 32 + (lane >> 4) * 16 + e;
  int Nn = dt * 16 + (lane & 15);
  out[p] = bfbits(W[((size_t)t * FDIM + K) * DDIM + Nn]);
}

__global__ void pack_node_kernel(const float* __restrict__ nt,  // [256][128]
                                 const float* __restrict__ ft,  // [256][128]
                                 unsigned short* __restrict__ out) {
  int p    = blockIdx.x * blockDim.x + threadIdx.x;  // 32768 total
  int e    = p & 15;
  int lane = (p >> 4) & 31;
  int frag = p >> 9;                                  // 0..63
  int ks = frag & 7, et = frag >> 3;
  int K  = ks * 32 + (lane >> 4) * 16 + e;
  int Nn = et * 16 + (lane & 15);
  size_t idx = (size_t)K * EDIM + Nn;
  out[p] = bfbits(nt[idx] + ft[idx]);                // fold node+feature trans
}

// ---------------------------------------------------------------------------
// Main fused kernel: 256 threads (8 wave32) per block, 16 nodes per block.
// ---------------------------------------------------------------------------
__global__ __launch_bounds__(256) void gatne_main_kernel(
    const float* __restrict__ targets,   // [N][256]
    const float* __restrict__ nbr,       // [N][4][10][256]
    const int*   __restrict__ types,     // [N]
    const float* __restrict__ w_tr,      // [4][64][128]
    const float* __restrict__ w_s1,      // [4][64][32]
    const float* __restrict__ w_s2,      // [4][32]
    const unsigned short* __restrict__ packEu,  // packed bf16 B frags (edge)
    const unsigned short* __restrict__ packNu,  // packed bf16 B frags (node)
    float* __restrict__ out)             // [N][128]
{
  __shared__ __bf16 agg_bf[2][NPB][APAD];   // bf16 A tiles, 2 types per round
  __shared__ __bf16 tgt_bf[NPB][APAD];      // bf16 A tile for node GEMM
  __shared__ float  eagg[NPB][TNUM][DDIM];  // edge_agg (fp32 WMMA C)
  __shared__ float  aemb[NPB][EDIM];        // accumulated embedding
  __shared__ float  eatt[NPB][DDIM];        // attention-weighted edge vector
  __shared__ float  spart[NPB][16];
  __shared__ float  att_s[NPB][TNUM];
  __shared__ float  scl[NPB];

  const int tid  = threadIdx.x;
  const int lane = tid & 31;
  const int wv   = tid >> 5;
  const int n0   = blockIdx.x * NPB;
  const v16bf* pE = (const v16bf*)packEu;
  const v16bf* pN = (const v16bf*)packNu;

  // ---- stage targets tile -> bf16 LDS (A-matrix rows) ----
  {
    int n = tid >> 4, c = tid & 15;
    const float* row = targets + (size_t)(n0 + n) * FDIM;
#pragma unroll
    for (int j = 0; j < 4; ++j) {
      int f = (c + 16 * j) * 4;
      float4 v = *(const float4*)(row + f);
      unsigned* dst = (unsigned*)&tgt_bf[n][f];
      dst[0] = pk2(v.x, v.y);
      dst[1] = pk2(v.z, v.w);
    }
  }

  // ---- edge aggregation (mean over s, hoisted) + bf16 WMMA GEMM ----
  for (int tp = 0; tp < 2; ++tp) {        // two edge types per round
    __syncthreads();                      // agg_bf free for reuse
    {
      int tl = tid >> 7;                  // type within pair
      int nn = (tid >> 3) & 15;           // node
      int f8 = tid & 7;                   // float4 chunk group
      const float* base =
          nbr + ((size_t)(n0 + nn) * TNUM + (tp * 2 + tl)) * SNUM * FDIM;
      float4 acc[8];
#pragma unroll
      for (int j = 0; j < 8; ++j) acc[j] = make_float4(0.f, 0.f, 0.f, 0.f);
      for (int s = 0; s < SNUM; ++s) {
        const float* rp = base + (size_t)s * FDIM;
#pragma unroll
        for (int j = 0; j < 8; ++j) {
          float4 v = *(const float4*)(rp + (f8 + 8 * j) * 4);
          acc[j].x += v.x; acc[j].y += v.y; acc[j].z += v.z; acc[j].w += v.w;
        }
      }
#pragma unroll
      for (int j = 0; j < 8; ++j) {       // fold 1/NEIGHBOR_SAMPLES here
        int f = (f8 + 8 * j) * 4;
        unsigned* dst = (unsigned*)&agg_bf[tl][nn][f];
        dst[0] = pk2(acc[j].x * 0.1f, acc[j].y * 0.1f);
        dst[1] = pk2(acc[j].z * 0.1f, acc[j].w * 0.1f);
      }
    }
    __syncthreads();
    {
      int tl = wv >> 2;                   // wave -> type within pair
      int dt = wv & 3;                    // wave -> 16-wide d tile
      int t  = tp * 2 + tl;
      const __bf16* arow = &agg_bf[tl][lane & 15][0];
      int hsel = (lane >> 4) * 8;
      v8f c = {0.f, 0.f, 0.f, 0.f, 0.f, 0.f, 0.f, 0.f};
#pragma unroll
      for (int ks = 0; ks < 8; ++ks) {
        int kb = ks * 32;
        union { struct { v8bf lo, hi; } p; v16bf v; } ua;
        ua.p.lo = *(const v8bf*)(arow + kb + 2 * hsel);        // K: kb+h*16..+7
        ua.p.hi = *(const v8bf*)(arow + kb + 2 * hsel + 8);    // K: kb+h*16+8..+15
        v16bf b = pE[(size_t)((t * 4 + dt) * 8 + ks) * 32 + lane];
        c = __builtin_amdgcn_wmma_f32_16x16x32_bf16(
                false, ua.v, false, b, (short)0, c, false, false);
      }
      int cn = lane & 15, h2 = lane >> 4; // C: vgpr i -> M = i + 8*h
#pragma unroll
      for (int i = 0; i < 8; ++i) eagg[8 * h2 + i][t][dt * 16 + cn] = c[i];
    }
  }

  // ---- node+feature GEMM: targets @ (node_trans + feature_trans) ----
  {
    const __bf16* arow = &tgt_bf[lane & 15][0];
    int hsel = (lane >> 4) * 8;
    v8f c = {0.f, 0.f, 0.f, 0.f, 0.f, 0.f, 0.f, 0.f};
#pragma unroll
    for (int ks = 0; ks < 8; ++ks) {
      int kb = ks * 32;
      union { struct { v8bf lo, hi; } p; v16bf v; } ua;
      ua.p.lo = *(const v8bf*)(arow + kb + 2 * hsel);
      ua.p.hi = *(const v8bf*)(arow + kb + 2 * hsel + 8);
      v16bf b = pN[(size_t)(wv * 8 + ks) * 32 + lane];
      c = __builtin_amdgcn_wmma_f32_16x16x32_bf16(
              false, ua.v, false, b, (short)0, c, false, false);
    }
    int cn = lane & 15, h2 = lane >> 4;
#pragma unroll
    for (int i = 0; i < 8; ++i) aemb[8 * h2 + i][wv * 16 + cn] = c[i];
  }
  __syncthreads();

  // ---- attention: h = tanh(eagg @ w_s1), scores = h @ w_s2, softmax ----
  const int n  = tid >> 4;
  const int j  = tid & 15;
  const int ty = types[n0 + n];
  {
    int tt = j >> 2;
    int a0 = (j & 3) * 8;
    float hacc[8] = {0.f, 0.f, 0.f, 0.f, 0.f, 0.f, 0.f, 0.f};
    for (int d = 0; d < DDIM; ++d) {
      float ea = eagg[n][tt][d];
      const float* wr = w_s1 + ((size_t)ty * DDIM + d) * ADIM + a0;
#pragma unroll
      for (int a = 0; a < 8; ++a) hacc[a] += ea * wr[a];
    }
    float sp = 0.f;
    const float* w2 = w_s2 + ty * ADIM + a0;
#pragma unroll
    for (int a = 0; a < 8; ++a) sp += tanhf(hacc[a]) * w2[a];
    spart[n][j] = sp;
  }
  __syncthreads();
  if (j == 0) {
    float sc[4];
#pragma unroll
    for (int t = 0; t < 4; ++t)
      sc[t] = spart[n][4 * t] + spart[n][4 * t + 1] +
              spart[n][4 * t + 2] + spart[n][4 * t + 3];
    float m = fmaxf(fmaxf(sc[0], sc[1]), fmaxf(sc[2], sc[3]));
    float ex0 = __expf(sc[0] - m), ex1 = __expf(sc[1] - m);
    float ex2 = __expf(sc[2] - m), ex3 = __expf(sc[3] - m);
    float inv = 1.f / (ex0 + ex1 + ex2 + ex3);
    att_s[n][0] = ex0 * inv; att_s[n][1] = ex1 * inv;
    att_s[n][2] = ex2 * inv; att_s[n][3] = ex3 * inv;
  }
  __syncthreads();
  {
#pragma unroll
    for (int q = 0; q < 4; ++q) {
      int d = j * 4 + q;
      eatt[n][d] = att_s[n][0] * eagg[n][0][d] + att_s[n][1] * eagg[n][1][d] +
                   att_s[n][2] * eagg[n][2][d] + att_s[n][3] * eagg[n][3][d];
    }
  }
  __syncthreads();

  // ---- edge_att @ w_tr[type] (per-node gathered matrix), add, L2-norm ----
  {
    int e0 = j * 8;
    float acc[8] = {0.f, 0.f, 0.f, 0.f, 0.f, 0.f, 0.f, 0.f};
    for (int d = 0; d < DDIM; ++d) {
      float ea = eatt[n][d];
      const float* wr = w_tr + ((size_t)ty * DDIM + d) * EDIM + e0;
#pragma unroll
      for (int a = 0; a < 8; ++a) acc[a] += ea * wr[a];
    }
    float ss = 0.f;
#pragma unroll
    for (int a = 0; a < 8; ++a) {
      float v = aemb[n][e0 + a] + acc[a];
      aemb[n][e0 + a] = v;
      ss += v * v;
    }
    spart[n][j] = ss;
  }
  __syncthreads();
  if (j == 0) {
    float ss = 0.f;
#pragma unroll
    for (int q = 0; q < 16; ++q) ss += spart[n][q];
    scl[n] = rsqrtf(fmaxf(ss, 1e-12f));
  }
  __syncthreads();
  {
    int e0 = j * 8;
    float s = scl[n];
    float* op = out + (size_t)(n0 + n) * EDIM + e0;
    float4 o0 = make_float4(aemb[n][e0 + 0] * s, aemb[n][e0 + 1] * s,
                            aemb[n][e0 + 2] * s, aemb[n][e0 + 3] * s);
    float4 o1 = make_float4(aemb[n][e0 + 4] * s, aemb[n][e0 + 5] * s,
                            aemb[n][e0 + 6] * s, aemb[n][e0 + 7] * s);
    *(float4*)op = o0;
    *(float4*)(op + 4) = o1;
  }
}

extern "C" void kernel_launch(void* const* d_in, const int* in_sizes, int n_in,
                              void* d_out, int out_size, void* d_ws, size_t ws_size,
                              hipStream_t stream) {
  const float* targets = (const float*)d_in[0];
  const float* nbr     = (const float*)d_in[1];
  const int*   types   = (const int*)d_in[2];
  const float* edge_W  = (const float*)d_in[3];   // edge_embedding_trans
  const float* w_tr    = (const float*)d_in[4];   // trans_weights
  const float* w_s1    = (const float*)d_in[5];   // trans_weights_s1
  const float* w_s2    = (const float*)d_in[6];   // trans_weights_s2
  const float* ntw     = (const float*)d_in[7];   // node_trans
  const float* ftw     = (const float*)d_in[8];   // feature_trans
  float* out = (float*)d_out;

  const int nnodes = in_sizes[2];                 // N (65536)

  // workspace: packed bf16 B fragments (edge: 128 frags, node: 64 frags)
  unsigned short* packE = (unsigned short*)d_ws;          // 128*512*2 = 128KB
  unsigned short* packN = packE + 128 * 512;              //  64*512*2 =  64KB

  pack_edge_kernel<<<256, 256, 0, stream>>>(edge_W, packE);
  pack_node_kernel<<<128, 256, 0, stream>>>(ntw, ftw, packN);

  gatne_main_kernel<<<nnodes / NPB, 256, 0, stream>>>(
      targets, nbr, types, w_tr, w_s1, w_s2, packE, packN, out);
}
